// Head_62371515072548
// MI455X (gfx1250) — compile-verified, matrix operand order checked by reading, not run on hardware
//
#include <hip/hip_runtime.h>
#include <hip/hip_bf16.h>

typedef __attribute__((ext_vector_type(16))) _Float16 v16h;
typedef __attribute__((ext_vector_type(8)))  _Float16 v8h;
typedef __attribute__((ext_vector_type(8)))  float    v8f;
typedef __attribute__((ext_vector_type(4)))  unsigned int u32x4;
typedef __attribute__((ext_vector_type(8)))  int          i32x8;
typedef __attribute__((ext_vector_type(4)))  int          i32x4;

union V16 { v16h v; v8h h[2]; };

__device__ inline v8h ld8h(const _Float16* p) { return *(const v8h*)p; }

// LDS byte offset of a __shared__ object (generic ptr low 32 bits = LDS offset)
__device__ inline unsigned int lds_off(const void* p) {
  return (unsigned int)(unsigned long long)p;
}

// Issue one TDM 2-D tile load: global (tile_d1 rows x tile_d0 contiguous f16
// elements, row stride = stride0 elements) -> LDS contiguous [d1][d0] f16.
// Descriptor per CDNA5 ISA. Tracked on TENSORcnt.
__device__ inline void tdm_load_2d_f16(unsigned int lds_byte_off,
                                       const _Float16* gptr,
                                       unsigned int tile_d0, unsigned int tile_d1,
                                       unsigned int stride0) {
  unsigned long long ga = (unsigned long long)gptr;
  u32x4 g0;
  g0.x = 1u;                                    // count=1 (valid user D#)
  g0.y = lds_byte_off;                          // lds_addr
  g0.z = (unsigned int)ga;                      // global_addr[31:0]
  g0.w = (unsigned int)((ga >> 32) & 0x1FFFFFFu) | (2u << 30); // addr[56:32]|type=2
  i32x8 g1;
  g1[0] = (1 << 16);                            // wg_mask=0, data_size=1 (2 bytes)
  g1[1] = (int)((tile_d0 & 0xFFFFu) << 16);     // tensor_dim0[15:0]
  g1[2] = (int)(((tile_d0 >> 16) & 0xFFFFu) | ((tile_d1 & 0xFFFFu) << 16));
  g1[3] = (int)(((tile_d1 >> 16) & 0xFFFFu) | ((tile_d0 & 0xFFFFu) << 16)); // tile_dim0
  g1[4] = (int)(tile_d1 & 0xFFFFu);             // tile_dim1 (tile_dim2 = 0)
  g1[5] = (int)stride0;                         // tensor_dim0_stride[31:0]
  g1[6] = 0;
  g1[7] = 0;
  i32x4 z4 = {0, 0, 0, 0};
  i32x8 z8 = {0, 0, 0, 0, 0, 0, 0, 0};
  __builtin_amdgcn_tensor_load_to_lds(g0, g1, z4, z4, z8, 0);
}

// ---------------------------------------------------------------------------
// Kernel 0: one-time prep: Wt[n][k] f16, n in [0,192) = concat(q,k,v) columns,
// k in [0,384). Row-major with k contiguous -> B-layout friendly + TDM tiles.
// ---------------------------------------------------------------------------
__global__ __launch_bounds__(256) void prep_wt_kernel(
    const float* __restrict__ Wq, const float* __restrict__ Wk,
    const float* __restrict__ Wv, _Float16* __restrict__ wt)
{
  int idx = blockIdx.x * 256 + threadIdx.x;    // 0 .. 192*384-1
  if (idx >= 192 * 384) return;
  int n = idx % 192;                            // coalesced over n
  int k = idx / 192;
  const float* Wm = (n < 64) ? Wq : (n < 128) ? Wk : Wv;
  wt[(size_t)n * 384 + k] = (_Float16)Wm[(size_t)k * 64 + (n & 63)];
}

// ---------------------------------------------------------------------------
// Kernel 1: fused QKV projection, TDM-fed.
//   x [131072,384] f32 @ Wt -> q,k [131072,64] f16, vT [512,64,256] f16.
// 8 waves/WG, each owns a 16-row tile; per 32-k chunk wave0 TDM-loads the
// Wt tile [192][32] into double-buffered LDS while everyone computes 12 WMMAs.
// ---------------------------------------------------------------------------
__global__ __launch_bounds__(256) void proj_qkv_kernel(
    const float* __restrict__ x, const _Float16* __restrict__ wt,
    _Float16* __restrict__ qg, _Float16* __restrict__ kg,
    _Float16* __restrict__ vtg)
{
  __shared__ _Float16 WtS[2][192 * 32];   // 24 KB double buffer

  const int tid  = threadIdx.x;
  const int lane = tid & 31;
  const int wv   = tid >> 5;
  const int l15  = lane & 15;
  const int hi   = lane >> 4;

  const size_t rowbase = (size_t)blockIdx.x * 128 + (size_t)wv * 16;
  const float* xrow = x + (rowbase + l15) * 384;

  v8f acc[12];
#pragma unroll
  for (int i = 0; i < 12; ++i) acc[i] = {};

  if (wv == 0)
    tdm_load_2d_f16(lds_off(&WtS[0][0]), wt, 32u, 192u, 384u);

  for (int kc = 0; kc < 12; ++kc) {
    const int kb = kc * 32;
    if (wv == 0) {
      if (kc + 1 < 12) {
        tdm_load_2d_f16(lds_off(&WtS[(kc + 1) & 1][0]), wt + (kc + 1) * 32,
                        32u, 192u, 384u);
        __builtin_amdgcn_s_wait_tensorcnt((short)1);
      } else {
        __builtin_amdgcn_s_wait_tensorcnt((short)0);
      }
    }
    __syncthreads();

    // A tile: x rows (rowbase + l15), K = kb..kb+31, f32 -> f16 in A layout
    V16 a;
    {
      const float* p0 = xrow + kb + hi * 8;
      const float* p1 = xrow + kb + 16 + hi * 8;
#pragma unroll
      for (int i = 0; i < 8; ++i) {
        a.v[i]     = (_Float16)p0[i];
        a.v[8 + i] = (_Float16)p1[i];
      }
    }

    const _Float16* Wb = &WtS[kc & 1][0];
#pragma unroll
    for (int nt = 0; nt < 12; ++nt) {
      const _Float16* wr = Wb + (nt * 16 + l15) * 32;
      V16 bt;
      bt.h[0] = *(const v8h*)(wr + hi * 8);
      bt.h[1] = *(const v8h*)(wr + 16 + hi * 8);
      acc[nt] = __builtin_amdgcn_wmma_f32_16x16x32_f16(
          false, a.v, false, bt.v, (short)0, acc[nt], false, false);
    }
    __syncthreads();
  }

  // Write results. C layout: lane owns (m = r + 8*hi, n = l15) per VGPR r.
#pragma unroll
  for (int nt = 0; nt < 12; ++nt) {
    const int mat = nt >> 2;               // 0=q 1=k 2=v
    const int h   = (nt & 3) * 16 + l15;
#pragma unroll
    for (int r = 0; r < 8; ++r) {
      const size_t m = rowbase + r + hi * 8;     // global row (b*256 + t)
      const _Float16 val = (_Float16)acc[nt][r];
      if (mat == 0)      qg[m * 64 + h] = val;
      else if (mat == 1) kg[m * 64 + h] = val;
      else {
        const size_t bb = m >> 8, t = m & 255;
        vtg[(bb * 64 + h) * 256 + t] = val;      // vT [b][h][t]
      }
    }
  }
}

// ---------------------------------------------------------------------------
// Kernel 2: causal flash attention. Block = 4 waves = 4 consecutive q-tiles
// of ONE batch; they share K/V key groups, which wave0 TDM-prefetches into
// double-buffered LDS (K tile 32x64, V tile 64x32 from vT). Scores via WMMA,
// online softmax with xor-shuffle reductions, P relaid C->A via LDS, PV WMMA.
// ---------------------------------------------------------------------------
__global__ __launch_bounds__(128) void attn_kernel(
    const _Float16* __restrict__ qg, const _Float16* __restrict__ kg,
    const _Float16* __restrict__ vtg, float* __restrict__ out)
{
  __shared__ _Float16 Kls[2][32 * 64];  // 8 KB
  __shared__ _Float16 Vls[2][64 * 32];  // 8 KB
  __shared__ _Float16 Pb[4][16 * 32];   // 4 KB per-wave P staging

  const int lane = threadIdx.x & 31;
  const int wv   = threadIdx.x >> 5;
  const int b    = blockIdx.x >> 2;             // 4 blocks per batch
  const int qt0  = (blockIdx.x & 3) * 4;        // first q-tile of block
  const int qt   = qt0 + wv;
  const int l15  = lane & 15;
  const int hi   = lane >> 4;

  const _Float16* kb_ptr = kg  + (size_t)b * 256 * 64;
  const _Float16* vb_ptr = vtg + (size_t)b * 64 * 256;

  // Q tile (16x64) as two f16 A operands (K = 0..31, 32..63)
  const _Float16* qrow = qg + ((size_t)b * 256 + qt * 16 + l15) * 64;
  V16 a0, a1;
  a0.h[0] = ld8h(qrow + hi * 8);
  a0.h[1] = ld8h(qrow + 16 + hi * 8);
  a1.h[0] = ld8h(qrow + 32 + hi * 8);
  a1.h[1] = ld8h(qrow + 48 + hi * 8);

  v8f acc0 = {}, acc1 = {}, acc2 = {}, acc3 = {};
  float mrun[8], lrun[8];
#pragma unroll
  for (int r = 0; r < 8; ++r) { mrun[r] = -__builtin_inff(); lrun[r] = 0.0f; }

  const int myng  = (qt * 16 + 15) / 32 + 1;          // my causal group count
  const int ngmax = ((qt0 + 3) * 16 + 15) / 32 + 1;   // block-wide max

  if (wv == 0) {
    tdm_load_2d_f16(lds_off(&Kls[0][0]), kb_ptr, 64u, 32u, 64u);
    tdm_load_2d_f16(lds_off(&Vls[0][0]), vb_ptr, 32u, 64u, 256u);
  }

  for (int g = 0; g < ngmax; ++g) {
    const int kb = g * 32;
    if (wv == 0) {
      if (g + 1 < ngmax) {
        const int nb = (g + 1) & 1;
        tdm_load_2d_f16(lds_off(&Kls[nb][0]), kb_ptr + (size_t)(kb + 32) * 64,
                        64u, 32u, 64u);
        tdm_load_2d_f16(lds_off(&Vls[nb][0]), vb_ptr + kb + 32,
                        32u, 64u, 256u);
        __builtin_amdgcn_s_wait_tensorcnt((short)2);
      } else {
        __builtin_amdgcn_s_wait_tensorcnt((short)0);
      }
    }
    __syncthreads();

    if (g < myng) {     // wave-uniform: EXEC stays all-1s for WMMA
      const _Float16* Kb = &Kls[g & 1][0];
      const _Float16* Vb = &Vls[g & 1][0];

      // K B-tiles: key sub-tiles x head chunks, from LDS
      const _Float16* kr0 = Kb + (size_t)l15 * 64;
      const _Float16* kr1 = Kb + (size_t)(16 + l15) * 64;
      V16 b0a, b0b, b1a, b1b;
      b0a.h[0] = ld8h(kr0 + hi * 8);      b0a.h[1] = ld8h(kr0 + 16 + hi * 8);
      b0b.h[0] = ld8h(kr0 + 32 + hi * 8); b0b.h[1] = ld8h(kr0 + 48 + hi * 8);
      b1a.h[0] = ld8h(kr1 + hi * 8);      b1a.h[1] = ld8h(kr1 + 16 + hi * 8);
      b1b.h[0] = ld8h(kr1 + 32 + hi * 8); b1b.h[1] = ld8h(kr1 + 48 + hi * 8);

      v8f s0 = {}, s1 = {};
      s0 = __builtin_amdgcn_wmma_f32_16x16x32_f16(false, a0.v, false, b0a.v, (short)0, s0, false, false);
      s0 = __builtin_amdgcn_wmma_f32_16x16x32_f16(false, a1.v, false, b0b.v, (short)0, s0, false, false);
      s1 = __builtin_amdgcn_wmma_f32_16x16x32_f16(false, a0.v, false, b1a.v, (short)0, s1, false, false);
      s1 = __builtin_amdgcn_wmma_f32_16x16x32_f16(false, a1.v, false, b1b.v, (short)0, s1, false, false);

      // causal mask + online softmax (rows live in 16-lane halves)
      const int n0 = kb + l15, n1 = n0 + 16;
      float p0[8], p1[8], scale[8];
#pragma unroll
      for (int r = 0; r < 8; ++r) {
        const int mg = qt * 16 + r + hi * 8;
        float x0 = (n0 <= mg) ? s0[r] : -__builtin_inff();
        float x1 = (n1 <= mg) ? s1[r] : -__builtin_inff();
        float mx = fmaxf(x0, x1);
        mx = fmaxf(mx, __shfl_xor(mx, 1, 32));
        mx = fmaxf(mx, __shfl_xor(mx, 2, 32));
        mx = fmaxf(mx, __shfl_xor(mx, 4, 32));
        mx = fmaxf(mx, __shfl_xor(mx, 8, 32));
        const float mn = fmaxf(mrun[r], mx);
        const float sc = __expf(mrun[r] - mn);
        const float e0 = __expf(x0 - mn);
        const float e1 = __expf(x1 - mn);
        float rs = e0 + e1;
        rs += __shfl_xor(rs, 1, 32);
        rs += __shfl_xor(rs, 2, 32);
        rs += __shfl_xor(rs, 4, 32);
        rs += __shfl_xor(rs, 8, 32);
        lrun[r] = lrun[r] * sc + rs;
        mrun[r] = mn;
        scale[r] = sc; p0[r] = e0; p1[r] = e1;
      }
#pragma unroll
      for (int r = 0; r < 8; ++r) {
        acc0[r] *= scale[r]; acc1[r] *= scale[r];
        acc2[r] *= scale[r]; acc3[r] *= scale[r];
      }

      // P (C layout f32) -> LDS f16 [row][col] -> reload in A layout
      _Float16* pb = Pb[wv];
#pragma unroll
      for (int r = 0; r < 8; ++r) {
        const int row = r + hi * 8;
        pb[row * 32 + l15]      = (_Float16)p0[r];
        pb[row * 32 + 16 + l15] = (_Float16)p1[r];
      }
      asm volatile("s_wait_dscnt 0" ::: "memory");
      V16 ap;
      ap.h[0] = *(const v8h*)(pb + l15 * 32 + hi * 8);
      ap.h[1] = *(const v8h*)(pb + l15 * 32 + 16 + hi * 8);

      // PV: B tiles from LDS V (rows contiguous in k), 4 head tiles of 16
#pragma unroll
      for (int t = 0; t < 4; ++t) {
        const _Float16* vr = Vb + (size_t)(t * 16 + l15) * 32;
        V16 vb;
        vb.h[0] = ld8h(vr + hi * 8);
        vb.h[1] = ld8h(vr + 16 + hi * 8);
        v8f* accp = (t == 0) ? &acc0 : (t == 1) ? &acc1 : (t == 2) ? &acc2 : &acc3;
        *accp = __builtin_amdgcn_wmma_f32_16x16x32_f16(
            false, ap.v, false, vb.v, (short)0, *accp, false, false);
      }
    }
    __syncthreads();
  }

  // normalize: 1/l_row and the reference's post-softmax 1/sqrt(E) quirk
  const float inv_sqrt_e = 0.05103103630798288f;  // 1/sqrt(384)
  float* orow = out + ((size_t)b * 256 + qt * 16) * 64;
#pragma unroll
  for (int r = 0; r < 8; ++r) {
    const float nrm = inv_sqrt_e / lrun[r];
    const size_t row = (size_t)(r + hi * 8);
    orow[row * 64 + 0  + l15] = acc0[r] * nrm;
    orow[row * 64 + 16 + l15] = acc1[r] * nrm;
    orow[row * 64 + 32 + l15] = acc2[r] * nrm;
    orow[row * 64 + 48 + l15] = acc3[r] * nrm;
  }
}

extern "C" void kernel_launch(void* const* d_in, const int* in_sizes, int n_in,
                              void* d_out, int out_size, void* d_ws, size_t ws_size,
                              hipStream_t stream) {
  const float* x  = (const float*)d_in[0];   // [512,256,384]
  const float* Wq = (const float*)d_in[1];   // [384,64]
  const float* Wk = (const float*)d_in[2];
  const float* Wv = (const float*)d_in[3];
  float* out = (float*)d_out;                // [512,256,64]

  const size_t ROWS = (size_t)512 * 256;     // 131072
  _Float16* qg  = (_Float16*)d_ws;           // [ROWS,64]
  _Float16* kg  = qg + ROWS * 64;            // [ROWS,64]
  _Float16* vtg = kg + ROWS * 64;            // [512,64,256]
  _Float16* wt  = vtg + (size_t)512 * 64 * 256;  // [192,384]

  prep_wt_kernel<<<(192 * 384 + 255) / 256, 256, 0, stream>>>(Wq, Wk, Wv, wt);
  proj_qkv_kernel<<<1024, 256, 0, stream>>>(x, wt, qg, kg, vtg);
  attn_kernel<<<512 * 4, 128, 0, stream>>>(qg, kg, vtg, out);
}
